// GCNModel_53377853554878
// MI455X (gfx1250) — matrix operand order repaired and checked by reading, not compiled, not run on hardware
//
#include <hip/hip_runtime.h>
#include <hip/hip_bf16.h>

#define NN 100000
#define EE 3200000
#define DD 256
#define HH 256
#define GG 512

typedef __attribute__((ext_vector_type(16))) _Float16 v16h;
typedef __attribute__((ext_vector_type(8)))  _Float16 v8h;
typedef __attribute__((ext_vector_type(4)))  _Float16 v4h;
typedef __attribute__((ext_vector_type(8)))  float    v8f;

// Workspace layout (32-bit words), 64B-aligned sections:
//   agg   : [0, NN*DD)                  float, zeroed
//   deg   : [NN*DD, NN*DD+NN)           uint -> float norm (in-place)
//   pooled: [NN*DD+NN, +GG*HH)          float, init -inf
//   w1h   : next 32768 words            f16 copy of W1 (256x256)
//   a16   : next NN*DD/2 words          f16 A matrix (agg + norm^2*x)
#define AGG_W   ((size_t)NN * DD)
#define DEG_W   (AGG_W + NN)
#define POOL_W  (DEG_W + (size_t)GG * HH)          // = 25,831,072 (16-word aligned)
#define W1H_W   (POOL_W + 32768)
#define INIT_W  POOL_W

__global__ void __launch_bounds__(256) init_ws_k(unsigned* __restrict__ ws) {
    size_t i = (size_t)blockIdx.x * 256 + threadIdx.x;
    if (i < DEG_W)            ws[i] = 0u;          // agg + deg
    else if (i < POOL_W)      ws[i] = 0xFF800000u; // pooled = -inf
}

__global__ void __launch_bounds__(256) degree_k(const int* __restrict__ ei,
                                                unsigned* __restrict__ deg) {
    int e = blockIdx.x * 256 + threadIdx.x;
    if (e < EE) atomicAdd(&deg[ei[e]], 1u);
}

__global__ void __launch_bounds__(256) norm_k(unsigned* __restrict__ deg) {
    int n = blockIdx.x * 256 + threadIdx.x;
    if (n < NN) {
        float d = (float)deg[n];
        ((float*)deg)[n] = rsqrtf(d);   // deg==0 -> +inf, matches deg**-0.5
    }
}

// One wave per edge: lane l covers 8 contiguous channels (coalesced 1KB row).
__global__ void __launch_bounds__(256) agg_k(const float* __restrict__ x,
                                             const int*   __restrict__ ei,
                                             const float* __restrict__ norm,
                                             float*       __restrict__ agg) {
    int lane = threadIdx.x & 31;
    int e = blockIdx.x * 8 + (threadIdx.x >> 5);
    if (e >= EE) return;
    int src = ei[e];
    int dst = ei[EE + e];
    float coef = norm[src] * norm[dst];
    const float4* xv = (const float4*)(x + (size_t)dst * DD) + lane * 2;
    float4 x0 = xv[0];
    float4 x1 = xv[1];
    float* ap = agg + (size_t)src * DD + lane * 8;
    unsafeAtomicAdd(ap + 0, coef * x0.x);
    unsafeAtomicAdd(ap + 1, coef * x0.y);
    unsafeAtomicAdd(ap + 2, coef * x0.z);
    unsafeAtomicAdd(ap + 3, coef * x0.w);
    unsafeAtomicAdd(ap + 4, coef * x1.x);
    unsafeAtomicAdd(ap + 5, coef * x1.y);
    unsafeAtomicAdd(ap + 6, coef * x1.z);
    unsafeAtomicAdd(ap + 7, coef * x1.w);
}

// W1 f32 -> f16 copy (one-time; B operand needs no conversion in GEMM loop).
__global__ void __launch_bounds__(256) w1h_k(const float* __restrict__ W1,
                                             _Float16*    __restrict__ w1h) {
    int i = blockIdx.x * 256 + threadIdx.x;   // over HH*DD = 65536
    if (i < HH * DD) w1h[i] = (_Float16)W1[i];
}

// A16 = f16(agg + norm^2 * x)   (self-loop fused; hoists all A converts).
__global__ void __launch_bounds__(256) acvt_k(const float* __restrict__ agg,
                                              const float* __restrict__ x,
                                              const float* __restrict__ norm,
                                              _Float16*    __restrict__ a16) {
    size_t t = (size_t)blockIdx.x * 256 + threadIdx.x;   // over NN*64
    if (t >= (size_t)NN * 64) return;
    int row = (int)(t >> 6);
    size_t base = t * 4;
    float n2 = norm[row];
    n2 *= n2;
    float4 a4 = *(const float4*)(agg + base);
    float4 x4 = *(const float4*)(x + base);
    v4h o;
    o[0] = (_Float16)(a4.x + n2 * x4.x);
    o[1] = (_Float16)(a4.y + n2 * x4.y);
    o[2] = (_Float16)(a4.z + n2 * x4.z);
    o[3] = (_Float16)(a4.w + n2 * x4.w);
    *(v4h*)(a16 + base) = o;
}

// Fused GEMM + bias + relu + segment-max.
// Block = 256 threads = 8 waves; block tile = 32 rows x 256 cols.
// Wave w: rows R0..R0+15 (R0 = blk*32 + (w>=4?16:0)), cols (w&3)*64 .. +63.
// Inner loop is pure f16 loads + v_wmma (all converts pre-hoisted).
__global__ void __launch_bounds__(256) gemm_pool_k(const _Float16* __restrict__ a16,
                                                   const _Float16* __restrict__ w1h,
                                                   const float*    __restrict__ b1,
                                                   const int*      __restrict__ batch,
                                                   float*          __restrict__ pooled) {
    const int tid  = threadIdx.x;
    const int lane = tid & 31;
    const int w    = tid >> 5;
    const int lo   = lane & 15;
    const bool hi  = lane >= 16;
    const int hs   = hi ? 8 : 0;

    const int R0  = blockIdx.x * 32 + ((w >= 4) ? 16 : 0);
    const int C0  = (w & 3) * 64;
    const int row = R0 + lo;      // A-matrix row held by this lane

    const _Float16* arow = a16 + (size_t)row * DD;

    v8f acc[4] = {};

    for (int kb = 0; kb < DD; kb += 32) {
        // A fragment (16x32 f16): lane<16 holds K {kb..kb+7, kb+16..kb+23},
        // lane>=16 holds K {kb+8..+15, kb+24..+31}. Two contiguous 16B chunks.
        v8h a0 = *(const v8h*)(arow + kb + hs);
        v8h a1 = *(const v8h*)(arow + kb + 16 + hs);
        v16h a = __builtin_shufflevector(a0, a1,
                                         0, 1, 2, 3, 4, 5, 6, 7,
                                         8, 9, 10, 11, 12, 13, 14, 15);
        // B fragment (32x16 f16): lane holds column N=lo, same K set as A:
        // {kb+hs..+7, kb+16+hs..+7} of the f16 W1 row (B = W1^T, row-major).
        const int kst = kb + (hi ? 16 : 0);
#pragma unroll
        for (int t = 0; t < 4; ++t) {
            const _Float16* brow = w1h + (size_t)(C0 + t * 16 + lo) * DD;
            v8h b0 = *(const v8h*)(brow + kb + hs);
            v8h b1v = *(const v8h*)(brow + kb + 16 + hs);
            v16h b = __builtin_shufflevector(b0, b1v, 0, 1, 2, 3, 4, 5, 6, 7, 8, 9, 10, 11, 12, 13, 14, 15);
            acc[t] = __builtin_amdgcn_wmma_f32_16x16x32_f16(
                false, a, false, b, (short)0, acc[t], false, false);
        }
    }

    // Epilogue: bias + relu + segment-max via signed-int atomicMax
    // (values >= 0 after relu; pooled pre-initialized to -inf bits).
    int gidx[8];
#pragma unroll
    for (int r = 0; r < 8; ++r) gidx[r] = batch[R0 + r + hs];

    int* poolI = (int*)pooled;
#pragma unroll
    for (int t = 0; t < 4; ++t) {
        const int col = C0 + t * 16 + lo;
        const float bias = b1[col];
#pragma unroll
        for (int r = 0; r < 8; ++r) {
            float v = acc[t][r] + bias;
            v = fmaxf(v, 0.0f);
            atomicMax(&poolI[(size_t)gidx[r] * HH + col], __float_as_int(v));
        }
    }
}

__global__ void __launch_bounds__(256) final_k(const float* __restrict__ pooled,
                                               const float* __restrict__ W2,
                                               const float* __restrict__ b2,
                                               float*       __restrict__ out) {
    int g = blockIdx.x * 256 + threadIdx.x;
    if (g >= GG) return;
    float s = b2[0];
    const float* pr = pooled + (size_t)g * HH;
#pragma unroll 8
    for (int j = 0; j < HH; ++j) s += pr[j] * W2[j];
    out[g] = s;
}

extern "C" void kernel_launch(void* const* d_in, const int* in_sizes, int n_in,
                              void* d_out, int out_size, void* d_ws, size_t ws_size,
                              hipStream_t stream) {
    const float* x     = (const float*)d_in[0];
    const int*   ei    = (const int*)  d_in[1];   // [2, E]
    const int*   batch = (const int*)  d_in[2];   // [N]
    const float* W1    = (const float*)d_in[3];   // [H, D]
    const float* b1    = (const float*)d_in[4];   // [H]
    const float* W2    = (const float*)d_in[5];   // [C, H] = [1, 256]
    const float* b2    = (const float*)d_in[6];   // [1]
    float* out = (float*)d_out;

    unsigned* ws_u   = (unsigned*)d_ws;
    float*    agg    = (float*)d_ws;
    unsigned* deg    = ws_u + AGG_W;
    float*    normv  = (float*)deg;               // in-place reuse
    float*    pooled = (float*)(ws_u + DEG_W);
    _Float16* w1h    = (_Float16*)(ws_u + POOL_W);
    _Float16* a16    = (_Float16*)(ws_u + W1H_W);

    init_ws_k<<<(int)((INIT_W + 255) / 256), 256, 0, stream>>>(ws_u);
    degree_k <<<(EE + 255) / 256, 256, 0, stream>>>(ei, deg);
    norm_k   <<<(NN + 255) / 256, 256, 0, stream>>>(deg);
    w1h_k    <<<(HH * DD + 255) / 256, 256, 0, stream>>>(W1, w1h);
    agg_k    <<<EE / 8, 256, 0, stream>>>(x, ei, normv, agg);            // 400000 blocks
    acvt_k   <<<(int)(((size_t)NN * 64 + 255) / 256), 256, 0, stream>>>(agg, x, normv, a16);
    gemm_pool_k<<<NN / 32, 256, 0, stream>>>(a16, w1h, b1, batch, pooled); // 3125 blocks
    final_k  <<<(GG + 255) / 256, 256, 0, stream>>>(pooled, W2, b2, out);
}
